// MMGATLayer_31525059953123
// MI455X (gfx1250) — matrix operand order repaired and checked by reading, not compiled, not run on hardware
//
#include <hip/hip_runtime.h>
#include <hip/hip_bf16.h>
#include <math.h>

// ---------------------------------------------------------------------------
// MMGATLayer fused kernel for MI455X (gfx1250, wave32, WMMA)
// ---------------------------------------------------------------------------
// Sizes fixed by the reference:
#define NN     16384
#define DD     16
#define FIN    512
#define FOUT   512
#define RR     8
#define HID    64
#define GG     4            // nodes per workgroup
#define NTHR   256          // 8 wave32
#define ROWS   48           // 4*8 att rows + 4 h rows, padded to 3 row-tiles
#define AROWS  36
#define A_STRIDE 536        // bf16 elems; 1072B row stride: 16B aligned, bank-spread
#define Z_STRIDE 516        // f32 elems
#define RESIDUAL 0.12f
#define BN_EPS   1e-5f

typedef __attribute__((ext_vector_type(16))) __bf16 v16bf;
typedef __attribute__((ext_vector_type(8)))  __bf16 v8bf;
typedef __attribute__((ext_vector_type(8)))  float  v8f;

// LDS layout (bytes)
#define OFF_A    0                       // 48*536*2        = 51456
#define OFF_Z    51456                   // 48*516*4        = 99072
#define OFF_MSG  150528                  // 2*4*512*4       = 16384
#define OFF_SCP  166912                  // 64*4*4          = 1024
#define OFF_DOT  167936                  // 48*4            = 192
#define OFF_ALPH 168128                  // 4*8*4           = 128
#define OFF_PB   168256                  // 4*2*64*4        = 2048
#define OFF_WB   170304                  // 8*4             = 32
#define LDS_BYTES 170336

__device__ __forceinline__ float lrelu(float x) { return x > 0.0f ? x : 0.2f * x; }

__device__ __forceinline__ v16bf combine16(v8bf lo, v8bf hi) {
    v16bf r;
#pragma unroll
    for (int i = 0; i < 8; ++i) { r[i] = lo[i]; r[i + 8] = hi[i]; }
    return r;
}

// A fragment (16x32 bf16, M x K), per 16-bit A VGPR table:
// lane l<16: K 0..7 then 16..23 ; lane l>=16: K 8..15 then 24..31
__device__ __forceinline__ v16bf load_fragA(const __bf16* Ash, int rowTile, int kt, int lane) {
    const int r  = lane & 15;
    const int hi = lane >> 4;
    const __bf16* p = Ash + (size_t)(rowTile * 16 + r) * A_STRIDE + kt * 32 + hi * 8;
    v8bf c0 = *(const v8bf*)(p);
    v8bf c1 = *(const v8bf*)(p + 16);
    return combine16(c0, c1);
}

// B fragment (32x16 bf16, K x N) from pre-transposed Wt[c][k] (stride FIN):
// lane l<16: K 0..7 then 16..23 of col l ; lane l>=16: K 8..15 then 24..31 of col l-16
__device__ __forceinline__ v16bf load_fragB(const __bf16* Wt, int colBase, int kt, int lane) {
    const int c  = lane & 15;
    const int hi = lane >> 4;
    const __bf16* p = Wt + (size_t)(colBase + c) * FIN + kt * 32 + hi * 8;
    v8bf c0 = *(const v8bf*)(p);
    v8bf c1 = *(const v8bf*)(p + 16);
    return combine16(c0, c1);
}

// --------------------------- prep: W_fc -> bf16 transposed -------------------
__global__ __launch_bounds__(256) void prep_w_kernel(const float* __restrict__ W0,
                                                     const float* __restrict__ W1,
                                                     __bf16* __restrict__ Wt) {
    int idx = blockIdx.x * 256 + threadIdx.x;       // 0 .. 2*512*512-1
    int m   = idx >> 18;
    int rem = idx & 262143;
    int c   = rem >> 9;
    int k   = rem & 511;
    const float* W = m ? W1 : W0;
    Wt[idx] = (__bf16)W[k * FIN + c];               // Wt[m][c][k] = W[k][c]
}

// ------------------------------- fused main ---------------------------------
__global__ __launch_bounds__(NTHR) void mmgat_fused_kernel(
    const float* __restrict__ src_h, const float* __restrict__ src_h_img,
    const int*   __restrict__ rel_type,
    const float* __restrict__ h,     const float* __restrict__ h_img,
    const float* __restrict__ a0,    const float* __restrict__ a1,
    const __bf16* __restrict__ Wt,                      // [2][512][512] bf16 transposed
    const float* __restrict__ Wp1,   const float* __restrict__ bp1,
    const float* __restrict__ Wp2,   const float* __restrict__ bp2,
    const float* __restrict__ bn_g,  const float* __restrict__ bn_b,
    const float* __restrict__ bn_m,  const float* __restrict__ bn_v,
    float* __restrict__ out) {

    extern __shared__ char smem[];
    __bf16* Ash  = (__bf16*)(smem + OFF_A);
    float*  Zsh  = (float*)(smem + OFF_Z);
    float*  MSG  = (float*)(smem + OFF_MSG);
    float*  SCP  = (float*)(smem + OFF_SCP);
    float*  DOTV = (float*)(smem + OFF_DOT);
    float*  ALPH = (float*)(smem + OFF_ALPH);
    float*  PB   = (float*)(smem + OFF_PB);
    float*  WB   = (float*)(smem + OFF_WB);

    const int tid  = threadIdx.x;
    const int lane = tid & 31;
    const int wid  = tid >> 5;            // 0..7 : wave owns cols [wid*64, wid*64+64)
    const int n0   = blockIdx.x * GG;     // first node of this workgroup

    // zero pad rows (36..47) of A once; they feed unused C rows but keep WMMA clean
    for (int i = tid; i < (ROWS - AROWS) * A_STRIDE; i += NTHR)
        Ash[AROWS * A_STRIDE + i] = (__bf16)0.0f;

    for (int m = 0; m < 2; ++m) {
        const float* src  = m ? src_h_img : src_h;
        const float* hsrc = m ? h_img : h;
        const float* av   = m ? a1 : a0;
        const __bf16* Wtm = Wt + (size_t)m * FIN * FOUT;

        // ---- Phase 1: relation max-pool -> bf16 A panel (+ h rows) ----------
        {
            const int n  = tid >> 6;          // 0..3 (wave-uniform: 2 waves per node)
            const int q  = tid & 63;
            const int fb = q * 8;
            const int node = n0 + n;

            float mx[RR][8];
#pragma unroll
            for (int r = 0; r < RR; ++r)
#pragma unroll
                for (int i = 0; i < 8; ++i) mx[r][i] = -3.0e38f;
            int cnt0=0,cnt1=0,cnt2=0,cnt3=0,cnt4=0,cnt5=0,cnt6=0,cnt7=0;

            const int* relrow = rel_type + (size_t)node * DD;
            const float* sbase = src + ((size_t)node * DD) * FIN + fb;
#pragma unroll
            for (int d = 0; d < DD; ++d) {
                const float* sp = sbase + (size_t)d * FIN;
                float4 va = *(const float4*)sp;
                float4 vb = *(const float4*)(sp + 4);
                float v[8] = {va.x, va.y, va.z, va.w, vb.x, vb.y, vb.z, vb.w};
                int r = __builtin_amdgcn_readfirstlane(relrow[d]);   // scalar branch
                switch (r) {
#define CASE(R,C) case R: { _Pragma("unroll") for (int i=0;i<8;++i) mx[R][i]=fmaxf(mx[R][i],v[i]); C++; } break;
                    CASE(0,cnt0) CASE(1,cnt1) CASE(2,cnt2) CASE(3,cnt3)
                    CASE(4,cnt4) CASE(5,cnt5) CASE(6,cnt6) CASE(7,cnt7)
#undef CASE
                }
            }
            int cnts[RR] = {cnt0,cnt1,cnt2,cnt3,cnt4,cnt5,cnt6,cnt7};
#pragma unroll
            for (int r = 0; r < RR; ++r) {
                // non-matching slots contribute 0.0 to the max (unless all matched)
                if (cnts[r] < DD) {
#pragma unroll
                    for (int i = 0; i < 8; ++i) mx[r][i] = fmaxf(mx[r][i], 0.0f);
                }
                __bf16* arow = Ash + (size_t)(n * RR + r) * A_STRIDE + fb;
#pragma unroll
                for (int i = 0; i < 8; ++i) arow[i] = (__bf16)mx[r][i];
            }
            // h row (row 32+n)
            const float* hp = hsrc + (size_t)node * FIN + fb;
            float4 ha = *(const float4*)hp;
            float4 hb = *(const float4*)(hp + 4);
            float hvv[8] = {ha.x, ha.y, ha.z, ha.w, hb.x, hb.y, hb.z, hb.w};
            __bf16* hrow = Ash + (size_t)(32 + n) * A_STRIDE + fb;
#pragma unroll
            for (int i = 0; i < 8; ++i) hrow[i] = (__bf16)hvv[i];
        }
        __syncthreads();

        // ---- Phase 2: [48 x 512] = A @ W  via v_wmma_f32_16x16x32_bf16 ------
        v8f acc[3][4];
#pragma unroll
        for (int rt = 0; rt < 3; ++rt)
#pragma unroll
            for (int ct = 0; ct < 4; ++ct) acc[rt][ct] = (v8f)0.0f;

        const int colBase = wid * 64;
        for (int kt = 0; kt < FIN / 32; ++kt) {
            v16bf afrag[3], bfrag[4];
#pragma unroll
            for (int rt = 0; rt < 3; ++rt) afrag[rt] = load_fragA(Ash, rt, kt, lane);
#pragma unroll
            for (int ct = 0; ct < 4; ++ct) bfrag[ct] = load_fragB(Wtm, colBase + ct * 16, kt, lane);
#pragma unroll
            for (int rt = 0; rt < 3; ++rt)
#pragma unroll
                for (int ct = 0; ct < 4; ++ct)
                    acc[rt][ct] = __builtin_amdgcn_wmma_f32_16x16x32_bf16(
                        false, afrag[rt], false, bfrag[ct], (short)0, acc[rt][ct],
                        false, false);
        }

        // ---- Phase 3: leaky-relu + spill C to LDS z-panel -------------------
        {
            const int cN = lane & 15;
            const int hi = lane >> 4;
#pragma unroll
            for (int rt = 0; rt < 3; ++rt)
#pragma unroll
                for (int ct = 0; ct < 4; ++ct) {
                    const int col = colBase + ct * 16 + cN;
#pragma unroll
                    for (int v = 0; v < 8; ++v) {
                        const int row = rt * 16 + v + (hi ? 8 : 0);
                        Zsh[(size_t)row * Z_STRIDE + col] = lrelu(acc[rt][ct][v]);
                    }
                }
        }
        __syncthreads();

        // ---- Phase 4: attention scores, softmax over R ----------------------
        {
            const int row  = tid >> 2;          // 0..63 ; valid rows 0..35
            const int part = tid & 3;
            if (row < AROWS) {
                const int f0  = part * 128;
                const int sel = (row < 32) ? 0 : FOUT;   // h rows use a[512:]
                const float* zr = Zsh + (size_t)row * Z_STRIDE + f0;
                const float* ar = av + sel + f0;
                float s = 0.0f;
                for (int f = 0; f < 128; ++f) s += zr[f] * ar[f];
                SCP[row * 4 + part] = s;
            }
        }
        __syncthreads();
        if (tid < AROWS)
            DOTV[tid] = SCP[tid * 4 + 0] + SCP[tid * 4 + 1] + SCP[tid * 4 + 2] + SCP[tid * 4 + 3];
        __syncthreads();
        if (tid < GG) {
            const float hs = DOTV[32 + tid];
            float sc[RR], smax = -3.0e38f;
#pragma unroll
            for (int r = 0; r < RR; ++r) { sc[r] = lrelu(DOTV[tid * RR + r] + hs); smax = fmaxf(smax, sc[r]); }
            float ssum = 0.0f;
#pragma unroll
            for (int r = 0; r < RR; ++r) { sc[r] = __expf(sc[r] - smax); ssum += sc[r]; }
            const float inv = 1.0f / ssum;
#pragma unroll
            for (int r = 0; r < RR; ++r) ALPH[tid * RR + r] = sc[r] * inv;
        }
        __syncthreads();

        // ---- Phase 5: msg = sum_r alpha*src_z + RESIDUAL*h_z ----------------
        {
            const int n  = tid >> 6;
            const int fb = (tid & 63) * 8;
            float al[RR];
#pragma unroll
            for (int r = 0; r < RR; ++r) al[r] = ALPH[n * RR + r];
            float* mg = MSG + (size_t)(m * GG + n) * FOUT + fb;
#pragma unroll
            for (int i = 0; i < 8; ++i) {
                const int f = fb + i;
                float acc2 = RESIDUAL * Zsh[(size_t)(32 + n) * Z_STRIDE + f];
#pragma unroll
                for (int r = 0; r < RR; ++r)
                    acc2 += al[r] * Zsh[(size_t)(n * RR + r) * Z_STRIDE + f];
                mg[i] = acc2;
            }
        }
        __syncthreads();
    } // modality loop

    // ---- Phase 6: p = tanh(msg @ W_p1 + b_p1) ------------------------------
    {
        const int n  = tid >> 6;
        const int j  = (tid >> 5) & 1;
        const int hb = (tid & 31) * 2;
        const float* mg = MSG + (size_t)(j * GG + n) * FOUT;
        float a = bp1[hb], b = bp1[hb + 1];
        for (int f = 0; f < FOUT; ++f) {
            const float mv = mg[f];
            a += mv * Wp1[f * HID + hb];
            b += mv * Wp1[f * HID + hb + 1];
        }
        PB[(size_t)(n * 2 + j) * HID + hb]     = tanhf(a);
        PB[(size_t)(n * 2 + j) * HID + hb + 1] = tanhf(b);
    }
    __syncthreads();

    // ---- Phase 7: w = tanh(p @ W_p2 + b_p2) --------------------------------
    if (tid < GG * 2) {
        const float* pr = PB + (size_t)tid * HID;
        float a = bp2[0];
#pragma unroll
        for (int hh = 0; hh < HID; ++hh) a += pr[hh] * Wp2[hh];
        WB[tid] = tanhf(a);
    }
    __syncthreads();

    // ---- Phase 8: beta softmax over 2, combine, eval BatchNorm, store ------
    {
        const int n  = tid >> 6;
        const int fb = (tid & 63) * 8;
        const float w0 = WB[n * 2 + 0], w1 = WB[n * 2 + 1];
        const float mw = fmaxf(w0, w1);
        const float e0 = __expf(w0 - mw), e1 = __expf(w1 - mw);
        const float b0 = e0 / (e0 + e1), b1 = e1 / (e0 + e1);
        float* op = out + (size_t)(n0 + n) * FOUT + fb;
#pragma unroll
        for (int i = 0; i < 8; ++i) {
            const int f = fb + i;
            const float o = b0 * MSG[(size_t)n * FOUT + f] +
                            b1 * MSG[(size_t)(GG + n) * FOUT + f];
            const float inv = rsqrtf(bn_v[f] + BN_EPS);
            op[i] = (o - bn_m[f]) * inv * bn_g[f] + bn_b[f];
        }
    }
}

// ---------------------------------------------------------------------------
extern "C" void kernel_launch(void* const* d_in, const int* in_sizes, int n_in,
                              void* d_out, int out_size, void* d_ws, size_t ws_size,
                              hipStream_t stream) {
    const float* src_h     = (const float*)d_in[0];
    const float* src_h_img = (const float*)d_in[1];
    const int*   rel_type  = (const int*)  d_in[2];
    const float* h         = (const float*)d_in[3];
    const float* h_img     = (const float*)d_in[4];
    const float* W_fc0     = (const float*)d_in[5];
    const float* a0        = (const float*)d_in[6];
    const float* W_fc1     = (const float*)d_in[7];
    const float* a1        = (const float*)d_in[8];
    const float* W_p1      = (const float*)d_in[9];
    const float* b_p1      = (const float*)d_in[10];
    const float* W_p2      = (const float*)d_in[11];
    const float* b_p2      = (const float*)d_in[12];
    const float* bn_gamma  = (const float*)d_in[13];
    const float* bn_beta   = (const float*)d_in[14];
    const float* bn_mean   = (const float*)d_in[15];
    const float* bn_var    = (const float*)d_in[16];
    float* out = (float*)d_out;

    __bf16* Wt = (__bf16*)d_ws;   // 2 * 512 * 512 * 2B = 1 MB

    prep_w_kernel<<<(2 * FIN * FOUT) / 256, 256, 0, stream>>>(W_fc0, W_fc1, Wt);
    mmgat_fused_kernel<<<NN / GG, NTHR, LDS_BYTES, stream>>>(
        src_h, src_h_img, rel_type, h, h_img, a0, a1, Wt,
        W_p1, b_p1, W_p2, b_p2, bn_gamma, bn_beta, bn_mean, bn_var, out);
}
